// nojit_customLSTMCell_22488448761884
// MI455X (gfx1250) — matrix operand order repaired and checked by reading, not compiled, not run on hardware
//
#include <hip/hip_runtime.h>
#include <stdint.h>

// ---------------------------------------------------------------------------
// Fused LSTM cell for MI455X (gfx1250, wave32, WMMA + async-to-LDS staging).
// Kernel 1: pack fp32 [x|hx] and [Wx|Wh] -> bf16 in d_ws (one pass).
// Kernel 2: gates GEMM (M=8192,N=4x1024,K=2048) via v_wmma_f32_16x16x32_bf16,
//           double-buffered LDS fed by global_load_async_to_lds_b128,
//           fused sigmoid/tanh LSTM epilogue writing h and C.
// ---------------------------------------------------------------------------

typedef __attribute__((ext_vector_type(16))) __bf16 v16bf;
typedef __attribute__((ext_vector_type(8)))  float  v8f;

#define B_ROWS   8192
#define H_DIM    1024
#define K_DIM    2048
#define BM       128        // block rows
#define BN       64         // block h-columns (per gate)
#define KT       32         // K tile (one bf16 WMMA K-step)
#define LDT      40         // padded LDS K-stride in halves (80B: conflict-free b128)
#define NTHREADS 256        // 8 waves
#define NKIT     (K_DIM / KT)   // 64

union FragU { uint4 u[2]; v16bf v; };   // 32B = 16 bf16 = one WMMA operand per lane

__device__ __forceinline__ unsigned pack_bf16x2(float a, float b) {
    unsigned ua = __float_as_uint(a), ub = __float_as_uint(b);
    ua = (ua + 0x7FFFu + ((ua >> 16) & 1u)) >> 16;   // round-to-nearest-even
    ub = (ub + 0x7FFFu + ((ub >> 16) & 1u)) >> 16;
    return (ua & 0xFFFFu) | (ub << 16);
}

__device__ __forceinline__ float sigmoid_f(float x) { return 1.0f / (1.0f + __expf(-x)); }
__device__ __forceinline__ float tanh_f(float x) {
    float e = __expf(-2.0f * x);
    return (1.0f - e) / (1.0f + e);
}

// ---------------------------------------------------------------------------
// Pack kernel: rows 0..8191 -> A = [x|hx] (bf16), rows 8192..12287 -> W=[Wx|Wh]
// One float4 chunk (4 elems) per thread, k-contiguous for coalescing.
// ---------------------------------------------------------------------------
__global__ __launch_bounds__(NTHREADS)
void lstm_pack_bf16(const float* __restrict__ x,  const float* __restrict__ hx,
                    const float* __restrict__ Wx, const float* __restrict__ Wh,
                    unsigned short* __restrict__ wsA, unsigned short* __restrict__ wsW)
{
    int id  = blockIdx.x * NTHREADS + threadIdx.x;   // float4-chunk id
    int row = id >> 9;                                // 512 chunks per 2048-row
    int kc  = (id & 511) << 2;                        // element offset in K
    const float* src;
    unsigned short* dst;
    if (row < B_ROWS) {
        src = (kc < 1024) ? &x[(size_t)row * 1024 + kc]
                          : &hx[(size_t)row * 1024 + (kc - 1024)];
        dst = &wsA[(size_t)row * K_DIM + kc];
    } else {
        int wr = row - B_ROWS;
        src = (kc < 1024) ? &Wx[(size_t)wr * 1024 + kc]
                          : &Wh[(size_t)wr * 1024 + (kc - 1024)];
        dst = &wsW[(size_t)wr * K_DIM + kc];
    }
    float4 v = *(const float4*)src;
    uint2 p;
    p.x = pack_bf16x2(v.x, v.y);
    p.y = pack_bf16x2(v.z, v.w);
    *(uint2*)dst = p;
}

// ---------------------------------------------------------------------------
// GEMM + LSTM epilogue
// ---------------------------------------------------------------------------
__global__ __launch_bounds__(NTHREADS, 1)
void lstm_gemm_wmma(const unsigned short* __restrict__ wsA,
                    const unsigned short* __restrict__ wsW,
                    const float* __restrict__ cx,
                    const float* __restrict__ bx, const float* __restrict__ bh,
                    float* __restrict__ outH, float* __restrict__ outC)
{
    __shared__ __align__(16) unsigned short As[2][BM * LDT];       // 2 x 10KB
    __shared__ __align__(16) unsigned short Bs[2][4 * BN * LDT];   // 2 x 20KB

    const int tid  = threadIdx.x;
    const int lane = tid & 31;
    const int wid  = tid >> 5;
    const int wRow = wid >> 1;        // 0..3 (32 rows each)
    const int wCol = wid & 1;         // 0..1 (32 cols each)
    const int bN   = blockIdx.x;      // 0..15
    const int bM   = blockIdx.y;      // 0..63

    // LDS byte addresses for async-load destinations (low 32 bits of generic ptr)
    const unsigned asBase[2] = { (unsigned)(uintptr_t)&As[0][0],
                                 (unsigned)(uintptr_t)&As[1][0] };
    const unsigned bsBase[2] = { (unsigned)(uintptr_t)&Bs[0][0],
                                 (unsigned)(uintptr_t)&Bs[1][0] };

    v8f acc[4][2][2];
    const v8f vzero = {0.f, 0.f, 0.f, 0.f, 0.f, 0.f, 0.f, 0.f};
#pragma unroll
    for (int g = 0; g < 4; ++g)
#pragma unroll
        for (int i = 0; i < 2; ++i)
#pragma unroll
            for (int j = 0; j < 2; ++j)
                acc[g][i][j] = vzero;

    const int laneN = lane & 15;
    const int laneK = lane >> 4;
    const int aOff0 = (wRow * 32 + laneN) * LDT + (laneK << 3);
    const int aOff1 = aOff0 + 16 * LDT;

    // --- async stage of one K-tile into LDS buffer `buf` -------------------
    auto stage = [&](int buf, int kt) {
        const unsigned aB = asBase[buf];
        const unsigned bB = bsBase[buf];
        // A tile: 128 rows x 32 halves = 512 x 16B chunks, 2 per thread
#pragma unroll
        for (int i = 0; i < 2; ++i) {
            int id  = tid + i * NTHREADS;
            int row = id >> 2;
            int c   = id & 3;
            unsigned goff = (unsigned)((((bM * BM + row) * K_DIM) + kt * KT + c * 8) * 2);
            unsigned lds  = aB + (unsigned)((row * LDT + c * 8) * 2);
            asm volatile("global_load_async_to_lds_b128 %0, %1, %2"
                         :: "v"(lds), "v"(goff), "s"(wsA) : "memory");
        }
        // W tiles: 4 gates x 64 rows x 32 halves = 1024 x 16B chunks, 4 per thread
#pragma unroll
        for (int i = 0; i < 4; ++i) {
            int id = tid + i * NTHREADS;
            int g  = id >> 8;
            int jd = id & 255;
            int j  = jd >> 2;
            int c  = jd & 3;
            unsigned goff = (unsigned)(((g * H_DIM + bN * BN + j) * K_DIM + kt * KT + c * 8) * 2);
            unsigned lds  = bB + (unsigned)(((g * BN + j) * LDT + c * 8) * 2);
            asm volatile("global_load_async_to_lds_b128 %0, %1, %2"
                         :: "v"(lds), "v"(goff), "s"(wsW) : "memory");
        }
    };

    stage(0, 0);    // prologue: fill buffer 0

    for (int kt = 0; kt < NKIT; ++kt) {
        const int cur = kt & 1;
        if (kt + 1 < NKIT) {
            stage(cur ^ 1, kt + 1);                               // prefetch next tile
            asm volatile("s_wait_asynccnt 0x6" ::: "memory");     // current 6 done
        } else {
            asm volatile("s_wait_asynccnt 0x0" ::: "memory");
        }
        __syncthreads();

        // ---- 16 x v_wmma_f32_16x16x32_bf16 --------------------------------
        FragU a0, a1;
        a0.u[0] = *(const uint4*)&As[cur][aOff0];
        a0.u[1] = *(const uint4*)&As[cur][aOff0 + 16];
        a1.u[0] = *(const uint4*)&As[cur][aOff1];
        a1.u[1] = *(const uint4*)&As[cur][aOff1 + 16];

#pragma unroll
        for (int g = 0; g < 4; ++g) {
#pragma unroll
            for (int tn = 0; tn < 2; ++tn) {
                FragU bf;
                int bOff = (g * BN + wCol * 32 + tn * 16 + laneN) * LDT + (laneK << 3);
                bf.u[0] = *(const uint4*)&Bs[cur][bOff];
                bf.u[1] = *(const uint4*)&Bs[cur][bOff + 16];
                acc[g][0][tn] = __builtin_amdgcn_wmma_f32_16x16x32_bf16(
                    false, a0.v, false, bf.v, (short)0, acc[g][0][tn], false, false);
                acc[g][1][tn] = __builtin_amdgcn_wmma_f32_16x16x32_bf16(
                    false, a1.v, false, bf.v, (short)0, acc[g][1][tn], false, false);
            }
        }
        __syncthreads();   // protect buf cur before it gets re-staged
    }

    // ---- fused LSTM epilogue ---------------------------------------------
    // C/D layout: VGPR r, lanes 0-15 -> M=r, lanes 16-31 -> M=8+r; N = lane&15
    const int rowHalf = laneK * 8;
#pragma unroll
    for (int tm = 0; tm < 2; ++tm) {
        int rowBase = bM * BM + wRow * 32 + tm * 16 + rowHalf;
#pragma unroll
        for (int tn = 0; tn < 2; ++tn) {
            int col = bN * BN + wCol * 32 + tn * 16 + laneN;
            float bF = bx[0 * H_DIM + col] + bh[0 * H_DIM + col];
            float bI = bx[1 * H_DIM + col] + bh[1 * H_DIM + col];
            float bG = bx[2 * H_DIM + col] + bh[2 * H_DIM + col];
            float bO = bx[3 * H_DIM + col] + bh[3 * H_DIM + col];
            v8f aF = acc[0][tm][tn], aI = acc[1][tm][tn];
            v8f aG = acc[2][tm][tn], aO = acc[3][tm][tn];
#pragma unroll
            for (int r = 0; r < 8; ++r) {
                size_t idx = (size_t)(rowBase + r) * H_DIM + col;
                float f = sigmoid_f(aF[r] + bF);
                float i = sigmoid_f(aI[r] + bI);
                float g = tanh_f(aG[r] + bG);
                float o = sigmoid_f(aO[r] + bO);
                float c = f * cx[idx] + i * g;
                float h = o * tanh_f(c);
                outH[idx] = h;
                outC[idx] = c;
            }
        }
    }
}

extern "C" void kernel_launch(void* const* d_in, const int* in_sizes, int n_in,
                              void* d_out, int out_size, void* d_ws, size_t ws_size,
                              hipStream_t stream) {
    const float* x  = (const float*)d_in[0];
    const float* hx = (const float*)d_in[1];
    const float* cx = (const float*)d_in[2];
    const float* Wx = (const float*)d_in[3];
    const float* Wh = (const float*)d_in[4];
    const float* bx = (const float*)d_in[5];
    const float* bh = (const float*)d_in[6];

    unsigned short* wsA = (unsigned short*)d_ws;                       // 32 MB
    unsigned short* wsW = wsA + (size_t)B_ROWS * K_DIM;                // 16 MB

    float* outH = (float*)d_out;
    float* outC = outH + (size_t)B_ROWS * H_DIM;

    // pack: (8192 + 4096) rows x 512 float4-chunks / 256 threads
    int packBlocks = ((B_ROWS + 4 * H_DIM) * (K_DIM / 4)) / NTHREADS;  // 24576
    lstm_pack_bf16<<<packBlocks, NTHREADS, 0, stream>>>(x, hx, Wx, Wh, wsA, wsW);

    dim3 grid(H_DIM / BN, B_ROWS / BM);   // (16, 64)
    lstm_gemm_wmma<<<grid, NTHREADS, 0, stream>>>(wsA, wsW, cx, bx, bh, outH, outC);
}
